// Model_83751862272173
// MI455X (gfx1250) — matrix operand order repaired and checked by reading, not compiled, not run on hardware
//
#include <hip/hip_runtime.h>
#include <hip/hip_bf16.h>
#include <stdint.h>

#define NU 200000
#define NI 100000
#define DD 64
#define EE 2000000
#define BB 8192
#define TILE 256

typedef __attribute__((ext_vector_type(16))) _Float16 v16h;
typedef __attribute__((ext_vector_type(8)))  float    v8f;

// ---------------------------------------------------------------- degrees
__global__ void __launch_bounds__(256)
deg_kernel(const int* __restrict__ eu, const int* __restrict__ ei,
           float* __restrict__ du, float* __restrict__ di, int n)
{
    int t = blockIdx.x * 256 + threadIdx.x;
    if (t < n) {
        atomicAdd(&du[eu[t]], 1.0f);
        atomicAdd(&di[ei[t]], 1.0f);
    }
}

// ---------------------------------------------------------------- edge norms
__global__ void __launch_bounds__(256)
edgenorm_kernel(const int* __restrict__ eu, const int* __restrict__ ei,
                const float* __restrict__ du, const float* __restrict__ di,
                float* __restrict__ nrm, int n)
{
    int t = blockIdx.x * 256 + threadIdx.x;
    if (t < n) nrm[t] = rsqrtf(du[eu[t]] * di[ei[t]]);
}

// ---------------------------------------------------------------- scatter
// One block per 256-edge tile. Edge metadata (u, i, norm) is staged into LDS
// with CDNA5 async global->LDS copies issued by wave 0 (ASYNCcnt path), then
// 8 waves process 32 edges each: one wave per edge, float2 per lane, f32
// global atomics (no-return -> STOREcnt, fire-and-forget into L2).
__global__ void __launch_bounds__(256)
scatter_kernel(const int* __restrict__ eu, const int* __restrict__ ei,
               const float* __restrict__ nrm,
               const float* __restrict__ uh, const float* __restrict__ ih,
               float* __restrict__ new_uh, float* __restrict__ new_ih, int E)
{
    __shared__ int   sh_u[TILE];
    __shared__ int   sh_i[TILE];
    __shared__ float sh_n[TILE];

    const int tileBase = blockIdx.x * TILE;
    const int cnt = min(TILE, E - tileBase);

    if (cnt == TILE) {
        if (threadIdx.x < 32) {
            const uint32_t l    = threadIdx.x;
            const uint32_t voff = (uint32_t)(tileBase + (int)l * 4) * 4u;
            const uint32_t ldsU = (uint32_t)(size_t)&sh_u[l * 4];
            const uint32_t ldsI = (uint32_t)(size_t)&sh_i[l * 4];
            const uint32_t ldsN = (uint32_t)(size_t)&sh_n[l * 4];
            // 32 lanes x 16B = 128 entries per instruction; offset:512 advances
            // BOTH the global and LDS address (ISA Sec 4.4) for the second half.
            asm volatile("global_load_async_to_lds_b128 %0, %1, %2"
                         :: "v"(ldsU), "v"(voff), "s"(eu) : "memory");
            asm volatile("global_load_async_to_lds_b128 %0, %1, %2 offset:512"
                         :: "v"(ldsU), "v"(voff), "s"(eu) : "memory");
            asm volatile("global_load_async_to_lds_b128 %0, %1, %2"
                         :: "v"(ldsI), "v"(voff), "s"(ei) : "memory");
            asm volatile("global_load_async_to_lds_b128 %0, %1, %2 offset:512"
                         :: "v"(ldsI), "v"(voff), "s"(ei) : "memory");
            asm volatile("global_load_async_to_lds_b128 %0, %1, %2"
                         :: "v"(ldsN), "v"(voff), "s"(nrm) : "memory");
            asm volatile("global_load_async_to_lds_b128 %0, %1, %2 offset:512"
                         :: "v"(ldsN), "v"(voff), "s"(nrm) : "memory");
            asm volatile("s_wait_asynccnt 0" ::: "memory");
        }
    } else {
        for (int t = threadIdx.x; t < cnt; t += 256) {
            sh_u[t] = eu[tileBase + t];
            sh_i[t] = ei[tileBase + t];
            sh_n[t] = nrm[tileBase + t];
        }
    }
    __syncthreads();

    const int wv   = threadIdx.x >> 5;
    const int lane = threadIdx.x & 31;
    const int start = wv * 32;
    const int end   = min(cnt, start + 32);

    for (int t = start; t < end; ++t) {
        const int   u = sh_u[t];
        const int   i = sh_i[t];
        const float w = sh_n[t];

        const float2 xu = *((const float2*)(uh + (size_t)u * DD) + lane);
        const float2 xi = *((const float2*)(ih + (size_t)i * DD) + lane);

        float* di = new_ih + (size_t)i * DD + lane * 2;
        float* du = new_uh + (size_t)u * DD + lane * 2;
        atomicAdd(di,     w * xu.x);
        atomicAdd(di + 1, w * xu.y);
        atomicAdd(du,     w * xi.x);
        atomicAdd(du + 1, w * xi.y);
    }
}

// ---------------------------------------------------------------- normalize + accumulate
// One wave per 16-row tile. Squared row norms = diagonal of the 16x16 Gram
// matrix of the (16x64) tile, computed with two v_wmma_f32_16x16x32_f16
// accumulations (K = 0..31, 32..63). A/B both use contiguous K order
// (k = 16*half + s); diagonal extracted via one half-wave shuffle on the f32 C layout.
__global__ void __launch_bounds__(256)
normalize_acc_kernel(const float* __restrict__ hin, float* __restrict__ hout,
                     float* __restrict__ eacc, int numTiles, float alpha)
{
    const int wave = (blockIdx.x * 256 + threadIdx.x) >> 5;
    if (wave >= numTiles) return;            // wave-uniform: EXEC stays all-ones
    const int lane = threadIdx.x & 31;
    const int n    = lane & 15;              // row within tile == our N column
    const int half = lane >> 4;              // lane half selects K sub-ranges
    const int row  = wave * 16 + n;
    const float* hrow = hin + (size_t)row * DD;

    v8f c = {};
#pragma unroll
    for (int kk = 0; kk < DD; kk += 32) {
        // A (16x32 f16): lanes 0-15 hold K=[0..15], lanes 16-31 hold K=[16..31]
        // (contiguous, matching B).
        const float* pa0 = hrow + kk + 16 * half;
        const float* pa1 = hrow + kk + 16 * half + 8;
        // B (32x16 f16) = A^T of the same tile: lane holds column n = row n of h,
        // K rows 16*half .. 16*half+15 packed two-per-VGPR.
        const float* pb  = hrow + kk + 16 * half;
        v16h a, b;
#pragma unroll
        for (int j = 0; j < 4; ++j) {
            a[2 * j]     = (_Float16)pa0[2 * j];
            a[2 * j + 1] = (_Float16)pa0[2 * j + 1];
            a[2 * j + 8] = (_Float16)pa1[2 * j];
            a[2 * j + 9] = (_Float16)pa1[2 * j + 1];
        }
#pragma unroll
        for (int j = 0; j < 8; ++j) {
            b[2 * j]     = (_Float16)pb[2 * j];
            b[2 * j + 1] = (_Float16)pb[2 * j + 1];
        }
        c = __builtin_amdgcn_wmma_f32_16x16x32_f16(false, a, false, b,
                                                   (short)0, c, false, false);
    }

    // C/D layout: lane l, VGPR v -> M = v + 8*(l>=16), N = l&15. We need D[n,n].
    const int v = n & 7;
    float own = c[0];
    own = (v == 1) ? c[1] : own;
    own = (v == 2) ? c[2] : own;
    own = (v == 3) ? c[3] : own;
    own = (v == 4) ? c[4] : own;
    own = (v == 5) ? c[5] : own;
    own = (v == 6) ? c[6] : own;
    own = (v == 7) ? c[7] : own;
    const float other  = __shfl_xor(own, 16);
    const float normsq = ((n >> 3) == half) ? own : other;

    const float inv = 1.0f / fmaxf(sqrtf(normsq), 1e-12f);

    // Each lane normalizes 32 columns of its row and folds into the
    // running embedding: lane half selects cols [32*half, 32*half+32).
    const float* src  = hrow + 32 * half;
    float*       dsth = hout + (size_t)row * DD + 32 * half;
    float*       dste = eacc + (size_t)row * DD + 32 * half;
#pragma unroll
    for (int j = 0; j < 32; j += 4) {
        float4 x = *(const float4*)(src + j);
        float4 y;
        y.x = x.x * inv; y.y = x.y * inv; y.z = x.z * inv; y.w = x.w * inv;
        *(float4*)(dsth + j) = y;
        float4 e = *(float4*)(dste + j);
        e.x += y.x * alpha; e.y += y.y * alpha;
        e.z += y.z * alpha; e.w += y.w * alpha;
        *(float4*)(dste + j) = e;
    }
}

// ---------------------------------------------------------------- final gather
__global__ void __launch_bounds__(256)
gather_kernel(const float* __restrict__ src, const int* __restrict__ idx,
              float* __restrict__ dst, int B)
{
    int t = blockIdx.x * 256 + threadIdx.x;     // B*16 threads, float4 each
    if (t < B * 16) {
        const int r = t >> 4;
        const int cofs = (t & 15) * 4;
        *(float4*)(dst + (size_t)r * DD + cofs) =
            *(const float4*)(src + (size_t)idx[r] * DD + cofs);
    }
}

// ---------------------------------------------------------------- launch
extern "C" void kernel_launch(void* const* d_in, const int* in_sizes, int n_in,
                              void* d_out, int out_size, void* d_ws, size_t ws_size,
                              hipStream_t stream)
{
    const float* user_feat = (const float*)d_in[0];
    const float* item_feat = (const float*)d_in[1];
    const int*   edge_u    = (const int*)d_in[2];
    const int*   edge_i    = (const int*)d_in[3];
    const int*   users     = (const int*)d_in[4];
    const int*   pos_items = (const int*)d_in[5];
    const int*   neg_items = (const int*)d_in[6];
    float* out = (float*)d_out;

    float* ws     = (float*)d_ws;
    float* uh     = ws;                            // NU*64
    float* ih     = uh     + (size_t)NU * DD;      // NI*64
    float* new_uh = ih     + (size_t)NI * DD;      // NU*64   (contiguous with
    float* new_ih = new_uh + (size_t)NU * DD;      // NI*64    new_uh for memset)
    float* ue     = new_ih + (size_t)NI * DD;      // NU*64
    float* ie     = ue     + (size_t)NU * DD;      // NI*64
    float* deg_u  = ie     + (size_t)NI * DD;      // NU      (contiguous with
    float* deg_i  = deg_u  + NU;                   // NI       deg_i for memset)
    float* enrm   = deg_i  + NI;                   // EE

    // degrees + per-edge symmetric norms
    hipMemsetAsync(deg_u, 0, (size_t)(NU + NI) * sizeof(float), stream);
    deg_kernel<<<(EE + 255) / 256, 256, 0, stream>>>(edge_u, edge_i, deg_u, deg_i, EE);
    edgenorm_kernel<<<(EE + 255) / 256, 256, 0, stream>>>(edge_u, edge_i, deg_u, deg_i, enrm, EE);

    // layer-0 state and running embeddings start at the raw features
    const size_t ubytes = (size_t)NU * DD * sizeof(float);
    const size_t ibytes = (size_t)NI * DD * sizeof(float);
    hipMemcpyAsync(uh, user_feat, ubytes, hipMemcpyDeviceToDevice, stream);
    hipMemcpyAsync(ue, user_feat, ubytes, hipMemcpyDeviceToDevice, stream);
    hipMemcpyAsync(ih, item_feat, ibytes, hipMemcpyDeviceToDevice, stream);
    hipMemcpyAsync(ie, item_feat, ibytes, hipMemcpyDeviceToDevice, stream);

    const int numEdgeTiles = (EE + TILE - 1) / TILE;
    for (int k = 0; k < 3; ++k) {
        hipMemsetAsync(new_uh, 0, (size_t)(NU + NI) * DD * sizeof(float), stream);
        scatter_kernel<<<numEdgeTiles, 256, 0, stream>>>(
            edge_u, edge_i, enrm, uh, ih, new_uh, new_ih, EE);
        const float alpha = 1.0f / (float)(k + 1);
        normalize_acc_kernel<<<(NU / 16 + 7) / 8, 256, 0, stream>>>(
            new_uh, uh, ue, NU / 16, alpha);
        normalize_acc_kernel<<<(NI / 16 + 7) / 8, 256, 0, stream>>>(
            new_ih, ih, ie, NI / 16, alpha);
    }

    gather_kernel<<<(BB * 16 + 255) / 256, 256, 0, stream>>>(ue, users,     out,                       BB);
    gather_kernel<<<(BB * 16 + 255) / 256, 256, 0, stream>>>(ie, pos_items, out + (size_t)BB * DD,     BB);
    gather_kernel<<<(BB * 16 + 255) / 256, 256, 0, stream>>>(ie, neg_items, out + (size_t)2 * BB * DD, BB);
}
